// SpatiotemporalFlashAttention_84799834292929
// MI455X (gfx1250) — matrix operand order repaired and checked by reading, not compiled, not run on hardware
//
#include <hip/hip_runtime.h>
#include <hip/hip_bf16.h>
#include <math.h>

typedef __attribute__((ext_vector_type(16))) __bf16 v16bf;
typedef __attribute__((ext_vector_type(8)))  __bf16 v8bf;
typedef __attribute__((ext_vector_type(8)))  float  v8f;

#define H_NUM   16
#define SEQ_L   2048
#define SEQ_S   2048
#define DQK     96      // 64 feat + 8 coord + 24 zero pad  (3 x K32)
#define DV      64
#define QTILE   64      // rows of Q per workgroup (16 per wave)
#define SCHUNK  64      // columns of S per iteration
#define NCHUNK  (SEQ_S / SCHUNK)
#define NWAVES  4
#define KSTRIDE (SCHUNK + 8)   // 72 bf16 = 144 B LDS row stride
#define VSTRIDE (DV + 8)       // 72 bf16
#define KBUFB   (DQK * KSTRIDE * 2)     // bytes per K buffer (13824)
#define VBUFB   (SCHUNK * VSTRIDE * 2)  // bytes per V buffer (9216)

__device__ __forceinline__ __bf16 f2bf(float f) {
    unsigned u = __builtin_bit_cast(unsigned, f);
    unsigned r = (u + 0x7FFFu + ((u >> 16) & 1u)) >> 16;
    unsigned short s = (unsigned short)r;
    return __builtin_bit_cast(__bf16, s);
}

__device__ __forceinline__ v16bf combine16(v8bf lo, v8bf hi) {
    v16bf r;
#pragma unroll
    for (int i = 0; i < 8; ++i) { r[i] = lo[i]; r[8 + i] = hi[i]; }
    return r;
}

// Low 32 bits of a flat pointer into __shared__ are the LDS byte offset (ISA §10.2).
__device__ __forceinline__ unsigned lds_off(const void* p) {
    return (unsigned)(unsigned long long)p;
}

// Async global -> LDS, 16 bytes per lane, tracked by ASYNCcnt (ISA §10 / §15.18.3).
__device__ __forceinline__ void async_copy_b128(unsigned lds_byte_off, const void* gptr) {
    asm volatile("global_load_async_to_lds_b128 %0, %1, off"
                 :: "v"(lds_byte_off), "v"(gptr)
                 : "memory");
}
__device__ __forceinline__ void wait_async0() {
    asm volatile("s_wait_asynccnt 0" ::: "memory");
}

// ---------------- pack kernels: f32 -> bf16, fold scales/alpha into Q ----------------

__global__ void pack_q(const float* __restrict__ qf, const float* __restrict__ qc,
                       const float* __restrict__ alpha, __bf16* __restrict__ Qp,
                       int B) {
    int t = blockIdx.x * blockDim.x + threadIdx.x;
    int total = B * H_NUM * SEQ_L;
    if (t >= total) return;
    int l = t % SEQ_L;
    int h = (t / SEQ_L) % H_NUM;
    int b = t / (SEQ_L * H_NUM);
    const float inv_sf = 0.125f;                 // 1/sqrt(64)
    const float inv_sc = 0.35355339059327373f;   // 1/sqrt(8)
    float ac = alpha[h] * inv_sc;
    const float* qfr = qf + ((size_t)b * SEQ_L + l) * 1024 + h * 64;
    const float* qcr = qc + ((size_t)b * SEQ_L + l) * 128  + h * 8;
    __bf16* dst = Qp + (size_t)t * DQK;
#pragma unroll 8
    for (int j = 0; j < 64; ++j) dst[j] = f2bf(qfr[j] * inv_sf);
#pragma unroll
    for (int j = 0; j < 8;  ++j) dst[64 + j] = f2bf(qcr[j] * ac);
#pragma unroll
    for (int j = 0; j < 24; ++j) dst[72 + j] = f2bf(0.0f);
}

// K' written CHUNK-TRANSPOSED: per (bh, s-chunk) a contiguous [DQK][SCHUNK] d-major
// block, so the flash kernel can stage it to LDS with pure async b128 copies.
__global__ void pack_k(const float* __restrict__ kf, const float* __restrict__ kc,
                       __bf16* __restrict__ Kp, int B) {
    int t = blockIdx.x * blockDim.x + threadIdx.x;
    int total = B * H_NUM * SEQ_S;
    if (t >= total) return;
    int s = t % SEQ_S;
    int h = (t / SEQ_S) % H_NUM;
    int b = t / (SEQ_S * H_NUM);
    int sc = s / SCHUNK, sl = s % SCHUNK;
    const float* kfr = kf + ((size_t)b * SEQ_S + s) * 1024 + h * 64;
    const float* kcr = kc + ((size_t)b * SEQ_S + s) * 128  + h * 8;
    __bf16* dst = Kp + (((size_t)(b * H_NUM + h) * NCHUNK + sc) * DQK) * SCHUNK + sl;
#pragma unroll 8
    for (int j = 0; j < 64; ++j) dst[j * SCHUNK] = f2bf(kfr[j]);
#pragma unroll
    for (int j = 0; j < 8;  ++j) dst[(64 + j) * SCHUNK] = f2bf(kcr[j]);
#pragma unroll
    for (int j = 0; j < 24; ++j) dst[(72 + j) * SCHUNK] = f2bf(0.0f);
}

__global__ void pack_v(const float* __restrict__ v, __bf16* __restrict__ Vp, int B) {
    int t = blockIdx.x * blockDim.x + threadIdx.x;
    int total = B * H_NUM * SEQ_S;
    if (t >= total) return;
    int s = t % SEQ_S;
    int h = (t / SEQ_S) % H_NUM;
    int b = t / (SEQ_S * H_NUM);
    const float* vr = v + ((size_t)b * SEQ_S + s) * 1024 + h * 64;
    __bf16* dst = Vp + (size_t)t * DV;
#pragma unroll 8
    for (int j = 0; j < 64; ++j) dst[j] = f2bf(vr[j]);
}

// ------- flash attention: bf16 WMMA + double-buffered async LDS staging -------
// grid: (L/QTILE, B*H), block: 128 threads (4 waves), wave w owns Q rows [w*16, w*16+16)

__global__ __launch_bounds__(128)
void flash_attn(const __bf16* __restrict__ Qp, const __bf16* __restrict__ Kp,
                const __bf16* __restrict__ Vp, float* __restrict__ out, int B) {
    __shared__ __align__(16) __bf16 Kt[2][DQK][KSTRIDE];    // d-major K' chunk, x2
    __shared__ __align__(16) __bf16 Vs[2][SCHUNK][VSTRIDE]; // s-major V chunk, x2
    __shared__ __align__(16) __bf16 Ps[NWAVES][16][64];     // per-wave P staging (C->A)

    const int tid  = threadIdx.x;
    const int wave = tid >> 5;
    const int lane = tid & 31;
    const int lh   = lane & 15;        // N / M-sub index
    const int hi   = lane >> 4;        // 0: lanes 0-15, 1: lanes 16-31

    const int bh    = blockIdx.y;               // b*H + h
    const int b     = bh / H_NUM;
    const int h     = bh % H_NUM;
    const int qbase = blockIdx.x * QTILE + wave * 16;

    // ---- resident Q A-fragments: row M = lh, K-halves: [hi*8, +8) and [16+hi*8, +8)
    const __bf16* qrow = Qp + ((size_t)bh * SEQ_L + qbase + lh) * DQK;
    v16bf qa[3];
#pragma unroll
    for (int c = 0; c < 3; ++c) {
        v8bf lo = *(const v8bf*)(qrow + c * 32 + hi * 8);
        v8bf up = *(const v8bf*)(qrow + c * 32 + 16 + hi * 8);
        qa[c] = combine16(lo, up);
    }

    // per-lane softmax state: index r corresponds to row M = r + 8*hi
    float mrow[8], lrow[8];
    v8f   o[4];
#pragma unroll
    for (int r = 0; r < 8; ++r) { mrow[r] = -INFINITY; lrow[r] = 0.0f; }
#pragma unroll
    for (int ot = 0; ot < 4; ++ot) o[ot] = v8f{};

    const __bf16* kbh = Kp + (size_t)bh * NCHUNK * DQK * SCHUNK;
    const __bf16* vbh = Vp + (size_t)bh * SEQ_S * DV;

    // per-lane LDS destinations (buffer 0); buffer 1 = +KBUFB / +VBUFB bytes
    unsigned koff[6], voff[4];
#pragma unroll
    for (int i = 0; i < 6; ++i) {
        int idx = tid + i * 128;           // 0..767, 8 x b128 per 64-wide d-row
        koff[i] = lds_off(&Kt[0][idx >> 3][(idx & 7) * 8]);
    }
#pragma unroll
    for (int i = 0; i < 4; ++i) {
        int idx = tid + i * 128;           // 0..511, 8 x b128 per 64-wide s-row
        voff[i] = lds_off(&Vs[0][idx >> 3][(idx & 7) * 8]);
    }

    auto stage = [&](int chunk, int p) {
        const __bf16* kchunk = kbh + (size_t)chunk * DQK * SCHUNK;
        const __bf16* vchunk = vbh + (size_t)chunk * SCHUNK * DV;
        unsigned kadd = p ? (unsigned)KBUFB : 0u;
        unsigned vadd = p ? (unsigned)VBUFB : 0u;
#pragma unroll
        for (int i = 0; i < 6; ++i)
            async_copy_b128(koff[i] + kadd, kchunk + (tid + i * 128) * 8);
#pragma unroll
        for (int i = 0; i < 4; ++i)
            async_copy_b128(voff[i] + vadd, vchunk + (tid + i * 128) * 8);
    };

    stage(0, 0);   // prologue: chunk 0 into buffer 0

    for (int sc = 0; sc < NCHUNK; ++sc) {
        const int p = sc & 1;

        wait_async0();     // buffer p (issued last iteration) has landed in LDS
        __syncthreads();   // all waves' writes visible; all reads of buffer p^1 retired

        if (sc + 1 < NCHUNK) stage(sc + 1, p ^ 1);   // overlap with compute below
        if (sc + 2 < NCHUNK) {
            __builtin_prefetch((const void*)(kbh + (size_t)(sc + 2) * DQK * SCHUNK), 0, 0);
            __builtin_prefetch((const void*)(vbh + (size_t)(sc + 2) * SCHUNK * DV), 0, 0);
        }

        // ---- scores: 16x64 tile = 4 N-tiles x 3 K32 chunks of bf16 WMMA
        v8f acc[4];
#pragma unroll
        for (int nt = 0; nt < 4; ++nt) {
            v8f a = v8f{};
#pragma unroll
            for (int c = 0; c < 3; ++c) {
                // B fragment: column N = lh (s within tile), 16 elements = K (d index), table order
                v16bf bk;
#pragma unroll
                for (int i = 0; i < 8; ++i) {
                    bk[i]     = Kt[p][c * 32 + hi * 8 + i][nt * 16 + lh];
                    bk[8 + i] = Kt[p][c * 32 + 16 + hi * 8 + i][nt * 16 + lh];
                }
                a = __builtin_amdgcn_wmma_f32_16x16x32_bf16(
                        false, qa[c], false, bk,
                        (short)0, a, false, false);
            }
            acc[nt] = a;
        }

        // ---- online softmax update
        float scale[8], rsum[8];
#pragma unroll
        for (int r = 0; r < 8; ++r) {
            float rm = acc[0][r];
#pragma unroll
            for (int nt = 1; nt < 4; ++nt) rm = fmaxf(rm, acc[nt][r]);
#pragma unroll
            for (int off = 1; off < 16; off <<= 1)
                rm = fmaxf(rm, __shfl_xor(rm, off, 32));
            float mn = fmaxf(mrow[r], rm);
            scale[r] = __expf(mrow[r] - mn);
            mrow[r]  = mn;
            rsum[r]  = 0.0f;
        }
#pragma unroll
        for (int nt = 0; nt < 4; ++nt) {
#pragma unroll
            for (int r = 0; r < 8; ++r) {
                float pv = __expf(acc[nt][r] - mrow[r]);
                rsum[r] += pv;
                Ps[wave][r + 8 * hi][nt * 16 + lh] = f2bf(pv);
            }
        }
#pragma unroll
        for (int r = 0; r < 8; ++r) {
            float rs = rsum[r];
#pragma unroll
            for (int off = 1; off < 16; off <<= 1)
                rs += __shfl_xor(rs, off, 32);
            lrow[r] = lrow[r] * scale[r] + rs;
        }
#pragma unroll
        for (int ot = 0; ot < 4; ++ot)
#pragma unroll
            for (int r = 0; r < 8; ++r) o[ot][r] *= scale[r];

        // ---- O += P x V  (2 K32 chunks x 4 D-tiles); LDS RAW within wave is in-order
#pragma unroll
        for (int kc = 0; kc < 2; ++kc) {
            v8bf plo = *(const v8bf*)(&Ps[wave][lh][kc * 32 + hi * 8]);
            v8bf pup = *(const v8bf*)(&Ps[wave][lh][kc * 32 + 16 + hi * 8]);
            v16bf pa = combine16(plo, pup);
#pragma unroll
            for (int ot = 0; ot < 4; ++ot) {
                v16bf bv;
#pragma unroll
                for (int i = 0; i < 8; ++i) {
                    bv[i]     = Vs[p][kc * 32 + hi * 8 + i][ot * 16 + lh];
                    bv[8 + i] = Vs[p][kc * 32 + 16 + hi * 8 + i][ot * 16 + lh];
                }
                o[ot] = __builtin_amdgcn_wmma_f32_16x16x32_bf16(
                            false, pa, false, bv,
                            (short)0, o[ot], false, false);
            }
        }
    }

    // ---- normalize and write output (B, L, H*64) f32
#pragma unroll
    for (int r = 0; r < 8; ++r) {
        float inv_l = 1.0f / lrow[r];
        int grow = qbase + r + 8 * hi;
        float* orow = out + ((size_t)b * SEQ_L + grow) * (H_NUM * 64) + h * 64;
#pragma unroll
        for (int ot = 0; ot < 4; ++ot)
            orow[ot * 16 + lh] = o[ot][r] * inv_l;
    }
}

extern "C" void kernel_launch(void* const* d_in, const int* in_sizes, int n_in,
                              void* d_out, int out_size, void* d_ws, size_t ws_size,
                              hipStream_t stream) {
    const float* qf    = (const float*)d_in[0];
    const float* kf    = (const float*)d_in[1];
    const float* qc    = (const float*)d_in[2];
    const float* kc    = (const float*)d_in[3];
    const float* v     = (const float*)d_in[4];
    const float* alpha = (const float*)d_in[5];

    int B = in_sizes[0] / (SEQ_L * 1024);   // 2

    __bf16* Qp = (__bf16*)d_ws;
    __bf16* Kp = Qp + (size_t)B * H_NUM * SEQ_L * DQK;
    __bf16* Vp = Kp + (size_t)B * H_NUM * SEQ_S * DQK;

    int nrowsQ = B * H_NUM * SEQ_L;
    int nrowsK = B * H_NUM * SEQ_S;
    pack_q<<<(nrowsQ + 255) / 256, 256, 0, stream>>>(qf, qc, alpha, Qp, B);
    pack_k<<<(nrowsK + 255) / 256, 256, 0, stream>>>(kf, kc, Kp, B);
    pack_v<<<(nrowsK + 255) / 256, 256, 0, stream>>>(v, Vp, B);

    dim3 grid(SEQ_L / QTILE, B * H_NUM);
    flash_attn<<<grid, NWAVES * 32, 0, stream>>>(Qp, Kp, Vp, (float*)d_out, B);
}
